// EGNNLayer_38405597561234
// MI455X (gfx1250) — matrix operand order repaired and checked by reading, not compiled, not run on hardware
//
#include <hip/hip_runtime.h>
#include <hip/hip_bf16.h>
#include <stdint.h>

// EGNN layer for MI455X (gfx1250): B=4, N=256, H=128, E=16, EIN=273.
// Strategy: strength-reduce the edge MLP (per-node partial GEMMs precomputed),
// then a WMMA(bf16, f32-accum) edge pipeline that streams edge_features once.

#define Bc 4
#define Nc 256
#define Hc 128
#define Ec 16
#define JSPLIT 8        // j-range split across blocks
#define JPI 8           // j's per inner iteration (one 16x8 M-tile row block)
#define ITERS 4         // (N / JSPLIT) / JPI

typedef __attribute__((ext_vector_type(16))) __bf16    v16bf;
typedef __attribute__((ext_vector_type(8)))  float     v8f;
typedef __attribute__((ext_vector_type(4)))  uint32_t  u32x4;

union BF16Frag { v16bf v; uint32_t u[8]; u32x4 q[2]; };

// round-to-nearest-even f32 -> bf16 pack helpers
__device__ __forceinline__ uint32_t pkbf(float a, float b) {
  uint32_t ua = __float_as_uint(a), ub = __float_as_uint(b);
  ua += 0x7FFFu + ((ua >> 16) & 1u);
  ub += 0x7FFFu + ((ub >> 16) & 1u);
  return (ua >> 16) | (ub & 0xFFFF0000u);
}
__device__ __forceinline__ uint16_t f2bf(float a) {
  uint32_t ua = __float_as_uint(a);
  ua += 0x7FFFu + ((ua >> 16) & 1u);
  return (uint16_t)(ua >> 16);
}

// ---------------------------------------------------------------------------
// prep_w: build bf16, transposed ([N][K]) weight buffers for B-fragments.
//   Wst[128][32]  : rows = output n, cols = K: [We1[256,:] ; We1[257:273,:] ; 0pad]
//   W2t[128][128] : We2 transposed
// ---------------------------------------------------------------------------
__global__ void prep_w(const float* We1, const float* We2,
                       uint16_t* Wst, uint16_t* W2t) {
  int t = threadIdx.x;
  for (int idx = t; idx < 128 * 32; idx += blockDim.x) {
    int n = idx >> 5, k = idx & 31;
    float v = 0.f;
    if (k == 0)       v = We1[256 * Hc + n];
    else if (k <= 16) v = We1[(256 + k) * Hc + n];
    Wst[n * 32 + k] = f2bf(v);
  }
  for (int idx = t; idx < 128 * 128; idx += blockDim.x) {
    int n = idx >> 7, k = idx & 127;
    W2t[n * 128 + k] = f2bf(We2[k * Hc + n]);
  }
}

// ---------------------------------------------------------------------------
// prep_node: per-node partial GEMMs for the strength reduction.
//   NI[row] = node @ We1[0:128]   + be1
//   NJ[row] = node @ We1[128:256]
//   pi[row] = node . Wa[0:128],  pj[row] = node . Wa[128:256]
// ---------------------------------------------------------------------------
__global__ void prep_node(const float* NF, const float* We1, const float* be1,
                          const float* Wa, float* NI, float* NJ,
                          float* pi, float* pj) {
  int row = blockIdx.x;   // b*N + n
  int t = threadIdx.x;    // 0..127
  __shared__ float x[Hc];
  __shared__ float red[Hc];
  x[t] = NF[(size_t)row * Hc + t];
  __syncthreads();
  float a0 = be1[t], a1 = 0.f;
  for (int k = 0; k < Hc; ++k) {
    float xv = x[k];
    a0 = fmaf(xv, We1[k * Hc + t], a0);
    a1 = fmaf(xv, We1[(128 + k) * Hc + t], a1);
  }
  NI[(size_t)row * Hc + t] = a0;
  NJ[(size_t)row * Hc + t] = a1;
  red[t] = x[t] * Wa[t];
  __syncthreads();
  for (int s = 64; s; s >>= 1) { if (t < s) red[t] += red[t + s]; __syncthreads(); }
  if (t == 0) pi[row] = red[0];
  __syncthreads();
  red[t] = x[t] * Wa[128 + t];
  __syncthreads();
  for (int s = 64; s; s >>= 1) { if (t < s) red[t] += red[t + s]; __syncthreads(); }
  if (t == 0) pj[row] = red[0];
}

// ---------------------------------------------------------------------------
// edge_kernel: the big WMMA pipeline.
// Grid (JSPLIT, 16 i-tiles, B). 256 threads = 8 wave32.
// Wave w owns M-tile rows m=0..15 -> (i_local = 2w + (m>>3), j_local = m&7).
// C layout: vgpr r, lane half h -> row m = r + 8h (so j_local = r, i_local = 2w+h).
// ---------------------------------------------------------------------------
__global__ __launch_bounds__(256, 1)
void edge_kernel(const float* CO, const float* EF, const float* MASK,
                 const float* Wa, const float* ba,
                 const float* lns, const float* lnb, const float* be2,
                 const uint16_t* Wst, const uint16_t* W2t,
                 const float* NI, const float* NJ,
                 const float* piG, const float* pjG,
                 const int* EIdx,
                 float* outE, float* msgp, float* cdirp) {
  const int js = blockIdx.x, it = blockIdx.y, b = blockIdx.z;
  const int i0 = it * 16;
  const int t = threadIdx.x;
  const int w = t >> 5, l = t & 31, half = l >> 4, lo = l & 15;

  __shared__ float ciS[16][3];
  __shared__ float waeS[16];
  __shared__ float cdirS[16 * 3];
  __shared__ float wrowS[16][8];
  __shared__ float d2S[16][8];
  __shared__ uint16_t hbuf[8][16][128];   // per-wave layer-1 activations (bf16)
  __shared__ float scal[2];               // Wa[256], ba

  if (t < 16) waeS[t] = Wa[257 + t];
  if (t < 48) {
    ciS[t / 3][t % 3] = CO[((size_t)(b * Nc) + i0 + t / 3) * 3 + (t % 3)];
    cdirS[t] = 0.f;
  }
  if (t == 48) scal[0] = Wa[256];
  if (t == 49) scal[1] = ba[0];

  const int il = 2 * w + half;        // i_local for this lane's C rows
  const int iG = i0 + il;

  // persistent per-lane data: C-init node term, LN params, layer-2 bias,
  // and the layer-1 B-fragments (K=32, one per N-tile).
  float niv[8], slnv[8], blnv[8], b2v[8];
  BF16Frag b1[8];
#pragma unroll
  for (int nt = 0; nt < 8; ++nt) {
    int n = lo + 16 * nt;
    niv[nt] = NI[((size_t)(b * Nc) + iG) * Hc + n];
    slnv[nt] = lns[n]; blnv[nt] = lnb[n]; b2v[nt] = be2[n];
    const u32x4* p = (const u32x4*)(Wst + n * 32 + half * 8);
    b1[nt].q[0] = p[0]; b1[nt].q[1] = p[2];
  }
  float msgacc[8];
#pragma unroll
  for (int nt = 0; nt < 8; ++nt) msgacc[nt] = 0.f;
  __syncthreads();

  for (int iter = 0; iter < ITERS; ++iter) {
    const int j0 = js * 32 + iter * JPI;

    // ---- per-edge scalar stage: d2, attention weight, coord direction ----
    if (t < 128) {
      int ail = t >> 3, ajl = t & 7;
      int i = i0 + ail, j = j0 + ajl;
      float cjx = CO[((size_t)(b * Nc) + j) * 3 + 0];
      float cjy = CO[((size_t)(b * Nc) + j) * 3 + 1];
      float cjz = CO[((size_t)(b * Nc) + j) * 3 + 2];
      float dx = ciS[ail][0] - cjx, dy = ciS[ail][1] - cjy, dz = ciS[ail][2] - cjz;
      float d2 = dx * dx + dy * dy + dz * dz;
      d2S[ail][ajl] = d2;
      const float4* er = (const float4*)(EF + (((size_t)(b * Nc) + i) * Nc + j) * Ec);
      float4 eA = er[0], eB = er[1], eC = er[2], eD = er[3];
      if (iter + 1 < ITERS)
        __builtin_prefetch(EF + (((size_t)(b * Nc) + i) * Nc + (j + JPI)) * Ec, 0, 0);
      float ed = eA.x * waeS[0] + eA.y * waeS[1] + eA.z * waeS[2] + eA.w * waeS[3]
               + eB.x * waeS[4] + eB.y * waeS[5] + eB.z * waeS[6] + eB.w * waeS[7]
               + eC.x * waeS[8] + eC.y * waeS[9] + eC.z * waeS[10] + eC.w * waeS[11]
               + eD.x * waeS[12] + eD.y * waeS[13] + eD.z * waeS[14] + eD.w * waeS[15];
      float lg = piG[b * Nc + i] + pjG[b * Nc + j] + d2 * scal[0] + ed + scal[1];
      float att = 1.f / (1.f + __expf(-lg));
      float adj = (float)EIdx[((size_t)(b * Nc) + i) * Nc + j];
      float aw = att * adj;
      wrowS[ail][ajl] = aw * MASK[b * Nc + i] * MASK[b * Nc + j];
      float inv = 1.f / sqrtf(d2 + 1e-8f);
      atomicAdd(&cdirS[ail * 3 + 0], dx * inv * aw);
      atomicAdd(&cdirS[ail * 3 + 1], dy * inv * aw);
      atomicAdd(&cdirS[ail * 3 + 2], dz * inv * aw);
    }
    __syncthreads();

    // ---- layer 1: C = NI[i] + NJ[j] (bias folded) + [d2|e|0] @ Wsmall ----
    v8f c[8];
#pragma unroll
    for (int nt = 0; nt < 8; ++nt) {
#pragma unroll
      for (int r = 0; r < 8; ++r)
        c[nt][r] = niv[nt] + NJ[((size_t)(b * Nc) + j0 + r) * Hc + lo + 16 * nt];
    }
    // A fragment: row M = lo -> edge (2w + (lo>>3), lo&7); lane half selects K range.
    {
      int mIl = 2 * w + (lo >> 3), mJl = lo & 7;
      const float4* er =
          (const float4*)(EF + (((size_t)(b * Nc) + i0 + mIl) * Nc + (j0 + mJl)) * Ec);
      float4 eA = er[0], eB = er[1], eC = er[2], eD = er[3];
      float d2v = d2S[mIl][mJl];
      BF16Frag a;
      if (half == 0) {   // K = 0..7 and 16..23  ->  [d2,e0..e6] , [e15,0..]
        a.u[0] = pkbf(d2v, eA.x); a.u[1] = pkbf(eA.y, eA.z);
        a.u[2] = pkbf(eA.w, eB.x); a.u[3] = pkbf(eB.y, eB.z);
        a.u[4] = pkbf(eD.w, 0.f); a.u[5] = 0; a.u[6] = 0; a.u[7] = 0;
      } else {           // K = 8..15 and 24..31 ->  [e7..e14] , [0..]
        a.u[0] = pkbf(eB.w, eC.x); a.u[1] = pkbf(eC.y, eC.z);
        a.u[2] = pkbf(eC.w, eD.x); a.u[3] = pkbf(eD.y, eD.z);
        a.u[4] = 0; a.u[5] = 0; a.u[6] = 0; a.u[7] = 0;
      }
#pragma unroll
      for (int nt = 0; nt < 8; ++nt)
        c[nt] = __builtin_amdgcn_wmma_f32_16x16x32_bf16(
            false, a.v, false, b1[nt].v, (short)0, c[nt], false, false);
    }

    // ---- LayerNorm + ReLU in registers; stage bf16 activations in LDS ----
    float mean[8], rstd[8];
#pragma unroll
    for (int r = 0; r < 8; ++r) {
      float s = 0.f, q = 0.f;
#pragma unroll
      for (int nt = 0; nt < 8; ++nt) { float x = c[nt][r]; s += x; q += x * x; }
#pragma unroll
      for (int msk = 1; msk < 16; msk <<= 1) {
        s += __shfl_xor(s, msk, 32);
        q += __shfl_xor(q, msk, 32);
      }
      float mu = s * (1.f / 128.f);
      float var = q * (1.f / 128.f) - mu * mu;
      mean[r] = mu; rstd[r] = rsqrtf(var + 1e-6f);
    }
#pragma unroll
    for (int nt = 0; nt < 8; ++nt) {
      int n = lo + 16 * nt;
#pragma unroll
      for (int r = 0; r < 8; ++r) {
        float y = (c[nt][r] - mean[r]) * rstd[r];
        y = fmaf(y, slnv[nt], blnv[nt]);
        y = fmaxf(y, 0.f);
        hbuf[w][r + 8 * half][n] = f2bf(y);   // row-major, A-frag friendly
      }
    }

    // ---- layer 2: c2 = h @ We2 + be2 (4 K-steps) ----
    v8f c2[8];
#pragma unroll
    for (int nt = 0; nt < 8; ++nt) {
#pragma unroll
      for (int r = 0; r < 8; ++r) c2[nt][r] = b2v[nt];
    }
#pragma unroll
    for (int ks = 0; ks < 4; ++ks) {
      BF16Frag a2;
      a2.q[0] = *(const u32x4*)&hbuf[w][lo][32 * ks + half * 8];
      a2.q[1] = *(const u32x4*)&hbuf[w][lo][32 * ks + half * 8 + 16];
#pragma unroll
      for (int nt = 0; nt < 8; ++nt) {
        BF16Frag b2;
        const uint16_t* wp = W2t + (lo + 16 * nt) * 128 + 32 * ks + half * 8;
        b2.q[0] = *(const u32x4*)wp;
        b2.q[1] = *(const u32x4*)(wp + 16);
        c2[nt] = __builtin_amdgcn_wmma_f32_16x16x32_bf16(
            false, a2.v, false, b2.v, (short)0, c2[nt], false, false);
      }
    }

    // ---- write edge messages; accumulate attention-weighted node messages ----
    float wr[8];
#pragma unroll
    for (int r = 0; r < 8; ++r) wr[r] = wrowS[il][r];
#pragma unroll
    for (int nt = 0; nt < 8; ++nt) {
      int n = lo + 16 * nt;
      float acc = msgacc[nt];
#pragma unroll
      for (int r = 0; r < 8; ++r) {
        float v = c2[nt][r];
        outE[(((size_t)(b * Nc) + iG) * Nc + (j0 + r)) * Hc + n] = v;
        acc = fmaf(wr[r], v, acc);
      }
      msgacc[nt] = acc;
    }
    __syncthreads();   // wrowS/d2S/cdirS reused next iteration
  }

  // ---- per-block partial outputs ----
  size_t pb = (size_t)((b * 16 + it) * JSPLIT + js);
#pragma unroll
  for (int nt = 0; nt < 8; ++nt)
    msgp[pb * 2048 + il * 128 + lo + 16 * nt] = msgacc[nt];
  if (t < 48) cdirp[pb * 48 + t] = cdirS[t];
}

// ---------------------------------------------------------------------------
// reduce_kernel: sum j-split partials -> node_messages, coord_update_dir
// ---------------------------------------------------------------------------
__global__ void reduce_kernel(const float* msgp, const float* cdirp,
                              float* NM, float* CD) {
  int row = blockIdx.x, t = threadIdx.x;
  int bb = row >> 8, n = row & 255;
  int it = n >> 4, il = n & 15;
  size_t base = (size_t)(bb * 16 + it) * JSPLIT;
  float s = 0.f;
  for (int js = 0; js < JSPLIT; ++js)
    s += msgp[(base + js) * 2048 + il * 128 + t];
  NM[(size_t)row * Hc + t] = s;
  if (t < 3) {
    float sc = 0.f;
    for (int js = 0; js < JSPLIT; ++js)
      sc += cdirp[(base + js) * 48 + il * 3 + t];
    CD[row * 3 + t] = sc;
  }
}

// ---------------------------------------------------------------------------
// node_kernel: new_node = node + MLP2([node | node_msg]) ; 1 row per block
// ---------------------------------------------------------------------------
__global__ void node_kernel(const float* NF, const float* NM,
                            const float* Wn1, const float* bn1,
                            const float* lns, const float* lnb,
                            const float* Wn2, const float* bn2, float* outN) {
  int row = blockIdx.x, t = threadIdx.x;
  __shared__ float xin[256];
  __shared__ float hs[128];
  __shared__ float red[128];
  __shared__ float stat[2];
  xin[t] = NF[(size_t)row * Hc + t];
  xin[128 + t] = NM[(size_t)row * Hc + t];
  __syncthreads();
  float h = bn1[t];
  for (int k = 0; k < 256; ++k) h = fmaf(xin[k], Wn1[k * Hc + t], h);
  red[t] = h; __syncthreads();
  for (int s = 64; s; s >>= 1) { if (t < s) red[t] += red[t + s]; __syncthreads(); }
  if (t == 0) stat[0] = red[0] * (1.f / 128.f);
  __syncthreads();
  float mu = stat[0], d = h - mu;
  red[t] = d * d; __syncthreads();
  for (int s = 64; s; s >>= 1) { if (t < s) red[t] += red[t + s]; __syncthreads(); }
  if (t == 0) stat[1] = rsqrtf(red[0] * (1.f / 128.f) + 1e-6f);
  __syncthreads();
  float y = fmaf(d * stat[1], lns[t], lnb[t]);
  hs[t] = fmaxf(y, 0.f);
  __syncthreads();
  float o = bn2[t];
  for (int k = 0; k < 128; ++k) o = fmaf(hs[k], Wn2[k * Hc + t], o);
  outN[(size_t)row * Hc + t] = o + xin[t];   // residual
}

// ---------------------------------------------------------------------------
// coord_kernel: coord_msg = MLP2(node_msg) -> scalar; coords += dir*msg*mask
// ---------------------------------------------------------------------------
__global__ void coord_kernel(const float* NM, const float* CD, const float* CO,
                             const float* MASK,
                             const float* Wc1, const float* bc1,
                             const float* lns, const float* lnb,
                             const float* Wc2, const float* bc2, float* outC) {
  int row = blockIdx.x, t = threadIdx.x;
  __shared__ float x[128];
  __shared__ float red[128];
  __shared__ float stat[2];
  x[t] = NM[(size_t)row * Hc + t];
  __syncthreads();
  float h = bc1[t];
  for (int k = 0; k < 128; ++k) h = fmaf(x[k], Wc1[k * Hc + t], h);
  red[t] = h; __syncthreads();
  for (int s = 64; s; s >>= 1) { if (t < s) red[t] += red[t + s]; __syncthreads(); }
  if (t == 0) stat[0] = red[0] * (1.f / 128.f);
  __syncthreads();
  float mu = stat[0], d = h - mu;
  red[t] = d * d; __syncthreads();
  for (int s = 64; s; s >>= 1) { if (t < s) red[t] += red[t + s]; __syncthreads(); }
  if (t == 0) stat[1] = rsqrtf(red[0] * (1.f / 128.f) + 1e-6f);
  __syncthreads();
  float y = fmaf(d * stat[1], lns[t], lnb[t]);
  y = fmaxf(y, 0.f);
  red[t] = y * Wc2[t]; __syncthreads();
  for (int s = 64; s; s >>= 1) { if (t < s) red[t] += red[t + s]; __syncthreads(); }
  if (t < 3) {
    float cm = red[0] + bc2[0];
    outC[row * 3 + t] = CO[row * 3 + t] + CD[row * 3 + t] * cm * MASK[row];
  }
}

// ---------------------------------------------------------------------------
extern "C" void kernel_launch(void* const* d_in, const int* in_sizes, int n_in,
                              void* d_out, int out_size, void* d_ws, size_t ws_size,
                              hipStream_t stream) {
  (void)in_sizes; (void)n_in; (void)out_size; (void)ws_size;
  const float* NF   = (const float*)d_in[0];
  const float* CO   = (const float*)d_in[1];
  const float* EF   = (const float*)d_in[2];
  const float* MASK = (const float*)d_in[3];
  const float* We1  = (const float*)d_in[4];
  const float* be1  = (const float*)d_in[5];
  const float* lnes = (const float*)d_in[6];
  const float* lneb = (const float*)d_in[7];
  const float* We2  = (const float*)d_in[8];
  const float* be2  = (const float*)d_in[9];
  const float* Wa   = (const float*)d_in[10];
  const float* ba   = (const float*)d_in[11];
  const float* Wn1  = (const float*)d_in[12];
  const float* bn1  = (const float*)d_in[13];
  const float* lnns = (const float*)d_in[14];
  const float* lnnb = (const float*)d_in[15];
  const float* Wn2  = (const float*)d_in[16];
  const float* bn2  = (const float*)d_in[17];
  const float* Wc1  = (const float*)d_in[18];
  const float* bc1  = (const float*)d_in[19];
  const float* lncs = (const float*)d_in[20];
  const float* lncb = (const float*)d_in[21];
  const float* Wc2  = (const float*)d_in[22];
  const float* bc2  = (const float*)d_in[23];
  const int*   EIdx = (const int*)d_in[24];

  char* ws = (char*)d_ws;                       // ~5.7 MB used
  uint16_t* Wst  = (uint16_t*)(ws + 0);         //   8 KB
  uint16_t* W2t  = (uint16_t*)(ws + 8192);      //  32 KB
  float* NI    = (float*)(ws + 40960);          // 512 KB
  float* NJ    = (float*)(ws + 565248);         // 512 KB
  float* piG   = (float*)(ws + 1089536);        //   4 KB
  float* pjG   = (float*)(ws + 1093632);        //   4 KB
  float* msgp  = (float*)(ws + 1097728);        //   4 MB
  float* cdirp = (float*)(ws + 5292032);        //  96 KB
  float* NM    = (float*)(ws + 5390336);        // 512 KB
  float* CD    = (float*)(ws + 5914624);        //  12 KB

  float* outN = (float*)d_out;
  float* outC = outN + (size_t)Bc * Nc * Hc;
  float* outE = outC + (size_t)Bc * Nc * 3;

  prep_w<<<1, 256, 0, stream>>>(We1, We2, Wst, W2t);
  prep_node<<<Bc * Nc, 128, 0, stream>>>(NF, We1, be1, Wa, NI, NJ, piG, pjG);
  dim3 g(JSPLIT, 16, Bc);
  edge_kernel<<<g, 256, 0, stream>>>(CO, EF, MASK, Wa, ba, lnes, lneb, be2,
                                     Wst, W2t, NI, NJ, piG, pjG, EIdx,
                                     outE, msgp, cdirp);
  reduce_kernel<<<Bc * Nc, 128, 0, stream>>>(msgp, cdirp, NM, CD);
  node_kernel<<<Bc * Nc, 128, 0, stream>>>(NF, NM, Wn1, bn1, lnns, lnnb, Wn2, bn2, outN);
  coord_kernel<<<Bc * Nc, 128, 0, stream>>>(NM, CD, CO, MASK, Wc1, bc1,
                                            lncs, lncb, Wc2, bc2, outC);
}
